// TMDLayer_36816459661845
// MI455X (gfx1250) — compile-verified, matrix operand order checked by reading, not run on hardware
//
#include <hip/hip_runtime.h>
#include <hip/hip_bf16.h>

typedef __attribute__((ext_vector_type(2)))  float        v2f;
typedef __attribute__((ext_vector_type(8)))  float        v8f;
typedef __attribute__((ext_vector_type(16))) __bf16       v16bf;
typedef __attribute__((ext_vector_type(4)))  unsigned int u32x4;
typedef __attribute__((ext_vector_type(8)))  int          i32x8;
typedef __attribute__((ext_vector_type(4)))  int          i32x4;

constexpr int   B   = 8;
constexpr int   N   = 2048;
constexpr int   F   = 256;
constexpr int   L   = 16;
constexpr float EPS = 0.25f;

// xor-shuffle across lanes via LDS swizzle hardware (no LDS memory touched).
// masks 1/2/4/8 stay within each 16-lane half of the wave32.
#define SWIZ_XOR_F(v, m) \
  __int_as_float(__builtin_amdgcn_ds_swizzle(__float_as_int(v), (0x1f | ((m) << 10))))

__device__ __forceinline__ unsigned short f2bf(float f) {
  unsigned u = __float_as_uint(f);
  u += 0x7FFFu + ((u >> 16) & 1u);   // round-to-nearest-even
  return (unsigned short)(u >> 16);
}

// ---------------------------------------------------------------------------
// Kernel 1: per-row front end. One block (256 thr) per (b,n) row.
//   z = x @ proj_w^T + proj_b        [B,N,16]
//   sq = |z|^2                       [B,N]
//   pi = sigmoid(w2 . relu(w1 z+b1)) [B,N]
// ---------------------------------------------------------------------------
__global__ void k_front(const float* __restrict__ x,  const float* __restrict__ pw,
                        const float* __restrict__ pb, const float* __restrict__ w1,
                        const float* __restrict__ b1, const float* __restrict__ w2,
                        const float* __restrict__ b2,
                        float* __restrict__ z, float* __restrict__ sq,
                        float* __restrict__ piv) {
  const int row = blockIdx.x;       // 0 .. B*N-1
  const int t   = threadIdx.x;      // 0 .. 255
  __shared__ float xs[F];
  __shared__ float red[F];
  __shared__ float zs[L];

  float xv = x[(size_t)row * F + t];
  xs[t] = xv;
  __syncthreads();

  // 16 groups of 16 threads each compute one z[l]
  const int g = t >> 4, i = t & 15;
  float p = 0.f;
#pragma unroll
  for (int c = 0; c < 16; ++c) p += xs[i + 16 * c] * pw[g * F + i + 16 * c];
  red[t] = p;
  __syncthreads();
  if (i == 0) {
    float s = 0.f;
#pragma unroll
    for (int j = 0; j < 16; ++j) s += red[(g << 4) + j];
    s += pb[g];
    zs[g] = s;
    z[(size_t)row * L + g] = s;
  }
  __syncthreads();
  if (t == 0) {
    float s = 0.f;
#pragma unroll
    for (int l = 0; l < L; ++l) s += zs[l] * zs[l];
    sq[row] = s;
  }
  // pi MLP: each thread owns one hidden unit f = t
  float h = b1[t];
#pragma unroll
  for (int l = 0; l < L; ++l) h += zs[l] * w1[t * L + l];
  h = fmaxf(h, 0.f);
  red[t] = h * w2[t];
  __syncthreads();
  for (int off = 128; off > 0; off >>= 1) {
    if (t < off) red[t] += red[t + off];
    __syncthreads();
  }
  if (t == 0) {
    float v = red[0] + b2[0];
    piv[row] = 1.f / (1.f + __expf(-v));
  }
}

// ---------------------------------------------------------------------------
// Kernel 1b: xT[b][f][n] = bf16(x[b][n][f]).  LDS-tiled 32x32 transpose.
// Grid: B * (N/32) * (F/32) blocks of 256.
// ---------------------------------------------------------------------------
__global__ void k_xt(const float* __restrict__ x, unsigned short* __restrict__ xT) {
  const int blk   = blockIdx.x;
  const int ftile = blk & (F / 32 - 1);         // 8
  const int ntile = (blk >> 3) & (N / 32 - 1);  // 64
  const int b     = blk >> 9;
  __shared__ float tile[32][33];

  {
    const int tn  = threadIdx.x >> 3;        // 0..31 (n in tile)
    const int tf4 = (threadIdx.x & 7) * 4;   // 4 consecutive f
    const float* src =
        x + ((size_t)(b * N + ntile * 32 + tn)) * F + ftile * 32 + tf4;
    float4 v = *(const float4*)src;
    tile[tn][tf4 + 0] = v.x;
    tile[tn][tf4 + 1] = v.y;
    tile[tn][tf4 + 2] = v.z;
    tile[tn][tf4 + 3] = v.w;
  }
  __syncthreads();
  {
    const int f  = threadIdx.x >> 3;         // 0..31 (f in tile)
    const int n4 = (threadIdx.x & 7) * 4;    // 4 consecutive n
    ushort4 o;
    o.x = f2bf(tile[n4 + 0][f]);
    o.y = f2bf(tile[n4 + 1][f]);
    o.z = f2bf(tile[n4 + 2][f]);
    o.w = f2bf(tile[n4 + 3][f]);
    *(ushort4*)(xT + ((size_t)(b * F + ftile * 32 + f)) * N + ntile * 32 + n4) = o;
  }
}

// ---------------------------------------------------------------------------
// f32 16x16x4 fragment helper: per-lane chunk of a z row.
// ---------------------------------------------------------------------------
__device__ __forceinline__ void load_zfrag(const float* __restrict__ zrow, int hi,
                                           v2f fr[4]) {
#pragma unroll
  for (int j = 0; j < 4; ++j) {
    fr[j][0] = zrow[4 * j + 2 * hi];
    fr[j][1] = zrow[4 * j + 2 * hi + 1];
  }
}

// ---------------------------------------------------------------------------
// Kernel 2: q[b,n] = sum_m exp(-d2(n,m)).  Gram via WMMA f32 16x16x4.
// ---------------------------------------------------------------------------
__global__ void k_q(const float* __restrict__ z, const float* __restrict__ sq,
                    float* __restrict__ q) {
  const int lane   = threadIdx.x & 31;
  const int wave   = threadIdx.x >> 5;
  const int stripe = blockIdx.x * 8 + wave;      // 0 .. B*N/16-1
  const int b      = stripe >> 7;                // 128 stripes per batch
  const int n0     = (stripe & 127) << 4;
  const float* zb  = z  + (size_t)b * N * L;
  const float* sqb = sq + (size_t)b * N;
  const int col = lane & 15, hi = lane >> 4;

  v2f a[4];
  load_zfrag(zb + (size_t)(n0 + col) * L, hi, a);
  float sqn[8];
#pragma unroll
  for (int r = 0; r < 8; ++r) sqn[r] = sqb[n0 + r + 8 * hi];

  float acc[8];
#pragma unroll
  for (int r = 0; r < 8; ++r) acc[r] = 0.f;

  for (int m0 = 0; m0 < N; m0 += 16) {
    v2f bb[4];
    load_zfrag(zb + (size_t)(m0 + col) * L, hi, bb);
    v8f g = {};
#pragma unroll
    for (int j = 0; j < 4; ++j)
      g = __builtin_amdgcn_wmma_f32_16x16x4_f32(false, a[j], false, bb[j],
                                                (short)0, g, false, false);
    const float sqm = sqb[m0 + col];
#pragma unroll
    for (int r = 0; r < 8; ++r) {
      float d2 = fmaxf(sqn[r] + sqm - 2.f * g[r], 0.f);
      acc[r] += __expf(-d2 * (1.0f / (4.0f * EPS)));   // 4*EPS == 1
    }
  }
#pragma unroll
  for (int r = 0; r < 8; ++r) {
    float v = acc[r];
    v += SWIZ_XOR_F(v, 1);
    v += SWIZ_XOR_F(v, 2);
    v += SWIZ_XOR_F(v, 4);
    v += SWIZ_XOR_F(v, 8);
    acc[r] = v;
  }
  if (col == 0) {
#pragma unroll
    for (int r = 0; r < 8; ++r) q[(size_t)b * N + n0 + 8 * hi + r] = acc[r];
  }
}

// ---------------------------------------------------------------------------
// Kernel 3: column scale c = pi / q
// ---------------------------------------------------------------------------
__global__ void k_colscale(const float* __restrict__ piv, const float* __restrict__ qv,
                           float* __restrict__ cv, int n) {
  int i = blockIdx.x * blockDim.x + threadIdx.x;
  if (i < n) cv[i] = piv[i] / qv[i];
}

// ---------------------------------------------------------------------------
// TDM: issue a 2D tensor tile load  xT[b][0..255][m0..m0+31] -> LDS [f][k].
// D# per CDNA5 ISA ch.8: group0 = {flags(count=1), lds_addr, gaddr_lo,
// gaddr_hi|type=2}; group1 packs data_size=2B, tensor_dim0=N, tensor_dim1=F,
// tile_dim0=32, tile_dim1=F, tensor_dim0_stride=N. tile_dim2=0 -> 2D, so
// remaining descriptor groups are unused (zero).
// This toolchain's builtin takes 6 args: (g0 u32x4, g1 i32x8, i32x4, i32x4,
// i32x8, cpol).
// ---------------------------------------------------------------------------
__device__ __forceinline__ void tdm_load_xtile(const unsigned short* gsrc,
                                               unsigned lds_off) {
  unsigned long long ga = (unsigned long long)(uintptr_t)gsrc;
  unsigned ga_lo = (unsigned)__builtin_amdgcn_readfirstlane((int)(unsigned)ga);
  unsigned ga_hi = (unsigned)__builtin_amdgcn_readfirstlane((int)(unsigned)(ga >> 32));
  lds_off        = (unsigned)__builtin_amdgcn_readfirstlane((int)lds_off);

  u32x4 g0;
  g0[0] = 1u;                                   // count=1 (valid user D#)
  g0[1] = lds_off;                              // lds_addr
  g0[2] = ga_lo;                                // global_addr[31:0]
  g0[3] = (ga_hi & 0x1FFFFFFu) | (2u << 30);    // global_addr[56:32] | type=2

  i32x8 g1;
  g1[0] = 1 << 16;          // data_size = 2 bytes
  g1[1] = N << 16;          // tensor_dim0[15:0] (=2048) at bits 63:48
  g1[2] = F << 16;          // tensor_dim1[15:0] (=256)  at bits 95:80
  g1[3] = 32 << 16;         // tile_dim0 = 32            at bits 127:112
  g1[4] = F;                // tile_dim1 = 256           at bits 143:128
  g1[5] = N;                // tensor_dim0_stride[31:0] = 2048
  g1[6] = 0;
  g1[7] = 0;

  i32x4 gz4 = {0, 0, 0, 0};
  i32x8 gz8 = {0, 0, 0, 0, 0, 0, 0, 0};
  __builtin_amdgcn_tensor_load_to_lds(g0, g1, gz4, gz4, gz8, 0);
}

// ---------------------------------------------------------------------------
// Kernel 4: fused apply.
// Block = 256 thr (8 waves) owns 128 n-rows of one batch; streams m in steps
// of 32. x tiles arrive transposed-in-global via TDM into double-buffered LDS
// (TENSORcnt). Per step & wave: gram (WMMA f32) -> Ktilde -> bf16 A-layout via
// per-wave LDS bounce -> 16x WMMA f32_16x16x32_bf16 against the x tile.
// Epilogue: out = x*(1-dt) + (dt/EPS)*acc/row.
// ---------------------------------------------------------------------------
__global__ void __launch_bounds__(256)
k_apply(const float* __restrict__ z, const float* __restrict__ sq,
        const float* __restrict__ cv, const unsigned short* __restrict__ xT,
        const float* __restrict__ x, const float* __restrict__ dtp,
        float* __restrict__ out) {
  const int t    = threadIdx.x;
  const int lane = t & 31;
  const int wave = t >> 5;
  const int blk  = blockIdx.x;            // 0 .. B*(N/128)-1
  const int b    = blk >> 4;              // 16 blocks per batch
  const int n0   = ((blk & 15) << 7) + (wave << 4);

  __shared__ __attribute__((aligned(16))) unsigned short xls[2][F * 32]; // 2x16KB
  __shared__ __attribute__((aligned(16))) unsigned short kls[8][16 * 32]; // 8KB

  const float* zb  = z   + (size_t)b * N * L;
  const float* sqb = sq  + (size_t)b * N;
  const float* cb  = cv  + (size_t)b * N;
  const unsigned short* xTb = xT + (size_t)b * F * N;
  const int col = lane & 15, hi = lane >> 4;

  v2f a[4];
  load_zfrag(zb + (size_t)(n0 + col) * L, hi, a);
  float sqn[8];
#pragma unroll
  for (int r = 0; r < 8; ++r) sqn[r] = sqb[n0 + r + 8 * hi];

  v8f zero = {};
  v8f acc[16];
#pragma unroll
  for (int ft = 0; ft < 16; ++ft) acc[ft] = zero;
  float rs[8];
#pragma unroll
  for (int r = 0; r < 8; ++r) rs[r] = 0.f;

  const unsigned ldsb0 = (unsigned)(size_t)(void*)&xls[0][0];
  const unsigned ldsb1 = (unsigned)(size_t)(void*)&xls[1][0];

  constexpr int NSTEPS = N / 32;   // 64
  if (wave == 0) tdm_load_xtile(xTb, ldsb0);   // prefetch tile 0

  for (int i = 0; i < NSTEPS; ++i) {
    const int m0 = i * 32;
    if (wave == 0) {
      if (i + 1 < NSTEPS) {
        tdm_load_xtile(xTb + (size_t)(i + 1) * 32, ((i + 1) & 1) ? ldsb1 : ldsb0);
        __builtin_amdgcn_s_wait_tensorcnt(1);   // tile i complete, i+1 in flight
      } else {
        __builtin_amdgcn_s_wait_tensorcnt(0);
      }
    }
    __syncthreads();   // tile i visible to all waves
    const unsigned short* xbuf = xls[i & 1];

    // --- two 16x16 Ktilde subtiles -> bf16 into per-wave LDS (A-layout src) ---
#pragma unroll
    for (int s = 0; s < 2; ++s) {
      const int mm = m0 + s * 16;
      v2f bb[4];
      load_zfrag(zb + (size_t)(mm + col) * L, hi, bb);
      v8f g = {};
#pragma unroll
      for (int j = 0; j < 4; ++j)
        g = __builtin_amdgcn_wmma_f32_16x16x4_f32(false, a[j], false, bb[j],
                                                  (short)0, g, false, false);
      const float sqm = sqb[mm + col];
      const float cm  = cb[mm + col];
#pragma unroll
      for (int r = 0; r < 8; ++r) {
        float d2 = fmaxf(sqn[r] + sqm - 2.f * g[r], 0.f);
        float kc = __expf(-d2 * (1.0f / (4.0f * EPS))) * cm;
        rs[r] += kc;
        kls[wave][(r + 8 * hi) * 32 + s * 16 + col] = f2bf(kc);
      }
    }
    // --- read back Ktilde as bf16 A fragment (16x32) ---
    union { uint4 u[2]; v16bf v; } afr;
    afr.u[0] = *(const uint4*)&kls[wave][col * 32 + hi * 8];
    afr.u[1] = *(const uint4*)&kls[wave][col * 32 + 16 + hi * 8];

    // --- 16 f-tiles: B fragment (32x16) from TDM-staged x tile ---
#pragma unroll
    for (int ft = 0; ft < 16; ++ft) {
      const int f = ft * 16 + col;
      union { uint4 u[2]; v16bf v; } bfr;
      bfr.u[0] = *(const uint4*)&xbuf[f * 32 + hi * 8];
      bfr.u[1] = *(const uint4*)&xbuf[f * 32 + 16 + hi * 8];
      acc[ft] = __builtin_amdgcn_wmma_f32_16x16x32_bf16(false, afr.v, false, bfr.v,
                                                        (short)0, acc[ft],
                                                        false, false);
    }
    __syncthreads();   // all waves done with tile i before its buffer refills
  }

  // row sums of Ktilde -> per-row scale (broadcast to all lanes via xor reduce)
  const float dt = dtp[0];
  float scale[8];
#pragma unroll
  for (int r = 0; r < 8; ++r) {
    float v = rs[r];
    v += SWIZ_XOR_F(v, 1);
    v += SWIZ_XOR_F(v, 2);
    v += SWIZ_XOR_F(v, 4);
    v += SWIZ_XOR_F(v, 8);
    scale[r] = (dt / EPS) / (v + 1e-5f);
  }

  const float* xb = x   + (size_t)b * N * F;
  float*       ob = out + (size_t)b * N * F;
  const float  omdt = 1.f - dt;
#pragma unroll
  for (int ft = 0; ft < 16; ++ft) {
    const int f = ft * 16 + col;
#pragma unroll
    for (int r = 0; r < 8; ++r) {
      const int n = n0 + r + 8 * hi;
      const size_t idx = (size_t)n * F + f;
      ob[idx] = xb[idx] * omdt + acc[ft][r] * scale[r];
    }
  }
}

// ---------------------------------------------------------------------------
extern "C" void kernel_launch(void* const* d_in, const int* in_sizes, int n_in,
                              void* d_out, int out_size, void* d_ws, size_t ws_size,
                              hipStream_t stream) {
  const float* x  = (const float*)d_in[0];
  const float* pw = (const float*)d_in[1];
  const float* pb = (const float*)d_in[2];
  const float* w1 = (const float*)d_in[3];
  const float* b1 = (const float*)d_in[4];
  const float* w2 = (const float*)d_in[5];
  const float* b2 = (const float*)d_in[6];
  const float* dt = (const float*)d_in[7];
  float* out = (float*)d_out;

  // workspace carve (~9.7 MB)
  float* z   = (float*)d_ws;                       // B*N*L
  float* sqv = z   + (size_t)B * N * L;            // B*N
  float* piv = sqv + (size_t)B * N;                // B*N
  float* qv  = piv + (size_t)B * N;                // B*N
  float* cvv = qv  + (size_t)B * N;                // B*N
  unsigned short* xT = (unsigned short*)(cvv + (size_t)B * N);  // B*F*N bf16

  k_front<<<B * N, 256, 0, stream>>>(x, pw, pb, w1, b1, w2, b2, z, sqv, piv);
  k_xt<<<B * (N / 32) * (F / 32), 256, 0, stream>>>(x, xT);
  k_q<<<(B * N / 16) / 8, 256, 0, stream>>>(z, sqv, qv);
  k_colscale<<<(B * N + 255) / 256, 256, 0, stream>>>(piv, qv, cvv, B * N);
  k_apply<<<B * (N / 128), 256, 0, stream>>>(z, sqv, cvv, xT, x, dt, out);
}
